// TransformerBlockModule_78503412236863
// MI455X (gfx1250) — matrix operand order repaired and checked by reading, not compiled, not run on hardware
//
#include <hip/hip_runtime.h>
#include <cstdint>

// ---------------- types ----------------
typedef __bf16 bf16_t;
typedef __attribute__((ext_vector_type(16))) __bf16 v16bf;
typedef __attribute__((ext_vector_type(8)))  float  v8f;

// ---------------- bf16 helpers (pure bit ops) ----------
__device__ __forceinline__ float bf2f(bf16_t h) {
    unsigned short u = __builtin_bit_cast(unsigned short, h);
    unsigned int x = ((unsigned int)u) << 16;
    return __builtin_bit_cast(float, x);
}
__device__ __forceinline__ bf16_t f2bf(float f) {
    unsigned int u = __builtin_bit_cast(unsigned int, f);
    unsigned int r = u + 0x7FFFu + ((u >> 16) & 1u);   // round-to-nearest-even
    unsigned short h = (unsigned short)(r >> 16);
    return __builtin_bit_cast(bf16_t, h);
}

// ---------------- constants ----------------
#define CD    1024      // model dim D
#define CNH   16
#define CDH   64
#define CT    2048      // tokens = B*S
#define CS    1024
#define CE    8
#define CH    2048
#define CBH   32        // B*NH
#define NSLOTS 5120     // padded MoE slots
#define MAXRB  40       // max 128-row blocks across experts

#define LDSPITCH 40     // elements per LDS row (80B: 16B-aligned, bank stride 20)
#define BUFBYTES (128 * LDSPITCH * 2)

// ============================================================
// WMMA GEMM mainloop, 128x128 C tile, NT layout, async-DMA staged:
//   A: row-major [M,K] (optional row-gather via rowmap, -1 = skip row)
//   B: row-major [N,K]
// Staging: GLOBAL_LOAD_ASYNC_TO_LDS_B128 into double-buffered LDS
// (ASYNCcnt), s_wait_asynccnt + workgroup barrier per K-step.
// 8 waves: wave w owns rows [32*(w&3), +32) x cols [64*(w>>2), +64)
// => 2 A frags, 4 B frags, 8 back-to-back WMMAs per K-step.
// ============================================================
__device__ __forceinline__ void gemm_mainloop(
    const bf16_t* __restrict__ A, long lda,
    const bf16_t* __restrict__ B, long ldb,
    int K, int Mrem, int Nrem,
    const int* __restrict__ rowmap,
    v8f acc[2][4])
{
    __shared__ bf16_t As[2][128][LDSPITCH];
    __shared__ bf16_t Bs[2][128][LDSPITCH];

    const int tid = threadIdx.x;

    // ---- per-thread DMA assignment: row = tid>>1, 16-elem half = tid&1 ----
    const int ar  = tid >> 1;
    const int acb = (tid & 1) * 16;
    bool avalid;
    const bf16_t* asrc;
    if (rowmap) {
        int tk = rowmap[ar];
        avalid = (tk >= 0);
        asrc = A + (avalid ? (long)tk * lda : 0) + acb;
    } else {
        avalid = (ar < Mrem);
        asrc = A + (long)ar * lda + acb;
    }
    const bool bvalid = (ar < Nrem);
    const bf16_t* bsrc = B + (long)ar * ldb + acb;

    const unsigned AsBase = (unsigned)(unsigned long long)&As[0][0][0];
    const unsigned BsBase = (unsigned)(unsigned long long)&Bs[0][0][0];
    const unsigned rowOff = (unsigned)(ar * (LDSPITCH * 2) + acb * 2);

    // issue 4 async b128 DMAs (2 A + 2 B) for K-step kk into buffer b
    auto prefetch = [&](int b, int kk) {
        const unsigned bo = (unsigned)(b * BUFBYTES);
        if (avalid) {
            unsigned l = AsBase + bo + rowOff;
            unsigned long long g = (unsigned long long)(const void*)(asrc + kk);
            asm volatile("global_load_async_to_lds_b128 %0, %1, off"
                         :: "v"(l), "v"(g) : "memory");
            asm volatile("global_load_async_to_lds_b128 %0, %1, off"
                         :: "v"(l + 16u), "v"(g + 16ull) : "memory");
        }
        if (bvalid) {
            unsigned l = BsBase + bo + rowOff;
            unsigned long long g = (unsigned long long)(const void*)(bsrc + kk);
            asm volatile("global_load_async_to_lds_b128 %0, %1, off"
                         :: "v"(l), "v"(g) : "memory");
            asm volatile("global_load_async_to_lds_b128 %0, %1, off"
                         :: "v"(l + 16u), "v"(g + 16ull) : "memory");
        }
    };

    const int lane = tid & 31, wv = tid >> 5;
    const int hf = lane >> 4, lm = lane & 15;
    const int mrow0 = (wv & 3) * 32;      // wave's row block within tile
    const int ncol0 = (wv >> 2) * 64;     // wave's col block within tile

    prefetch(0, 0);

    for (int kk = 0; kk < K; kk += 32) {
        const int b = (kk >> 5) & 1;
        // my DMAs into buffer b are done; barrier publishes everyone's and
        // guarantees no wave still reads buffer b^1 before we overwrite it.
        asm volatile("s_wait_asynccnt 0x0" ::: "memory");
        __syncthreads();
        if (kk + 32 < K) prefetch(b ^ 1, kk + 32);

        // ---- A fragments (16x32 each): lane<16 K{0..7,16..23}, lane>=16 K{8..15,24..31}
        v16bf a0, a1;
        const int ar0 = mrow0 + lm;
#pragma unroll
        for (int e = 0; e < 8; ++e) {
            a0[e]     = As[b][ar0][hf * 8 + e];
            a0[8 + e] = As[b][ar0][16 + hf * 8 + e];
            a1[e]     = As[b][ar0 + 16][hf * 8 + e];
            a1[8 + e] = As[b][ar0 + 16][16 + hf * 8 + e];
        }
        // ---- B fragments (32x16 each): lane<16 K{0..7,16..23}, lane>=16 K{8..15,24..31}, lane&15 = N
        v16bf bf[4];
#pragma unroll
        for (int nt = 0; nt < 4; ++nt) {
            const int bn = ncol0 + nt * 16 + lm;
#pragma unroll
            for (int e = 0; e < 16; ++e) bf[nt][e] = Bs[b][bn][hf * 8 + e + (e >= 8 ? 8 : 0)];
        }
        // ---- 8 WMMAs back-to-back ----
#pragma unroll
        for (int nt = 0; nt < 4; ++nt) {
            acc[0][nt] = __builtin_amdgcn_wmma_f32_16x16x32_bf16(
                false, a0, false, bf[nt], (short)0, acc[0][nt], false, false);
            acc[1][nt] = __builtin_amdgcn_wmma_f32_16x16x32_bf16(
                false, a1, false, bf[nt], (short)0, acc[1][nt], false, false);
        }
        __syncthreads();
    }
}

// ============================================================
// Generic batched NT GEMM: off = (z/zdiv)*s1 + (z%zdiv)*s2 per tensor.
// Epilogue: val = alpha*acc + bias[col]; f32 (+resid) or bf16 store.
// ============================================================
__global__ __launch_bounds__(256)
void gemm_nt_batched(const bf16_t* __restrict__ A, long lda, long sA1, long sA2,
                     const bf16_t* __restrict__ B, long ldb, long sB1, long sB2,
                     const float* __restrict__ bias,
                     void* __restrict__ Cout, int cF32, long ldc, long sC1, long sC2,
                     const float* __restrict__ resid,
                     int M, int N, int K, int zdiv, float alpha)
{
    const int z = blockIdx.z;
    const long zq = z / zdiv, zr = z % zdiv;
    const bf16_t* Ab = A + zq * sA1 + zr * sA2 + (long)blockIdx.y * 128 * lda;
    const bf16_t* Bb = B + zq * sB1 + zr * sB2 + (long)blockIdx.x * 128 * ldb;

    v8f acc[2][4] = {};
    gemm_mainloop(Ab, lda, Bb, ldb, K,
                  M - blockIdx.y * 128, N - blockIdx.x * 128, nullptr, acc);

    const long Cbase = zq * sC1 + zr * sC2;
    const int tid = threadIdx.x, lane = tid & 31, wv = tid >> 5;
    const int hf = lane >> 4, lm = lane & 15;
    const int mrow0 = (wv & 3) * 32, ncol0 = (wv >> 2) * 64;
#pragma unroll
    for (int am = 0; am < 2; ++am)
#pragma unroll
        for (int nt = 0; nt < 4; ++nt) {
            const int col = blockIdx.x * 128 + ncol0 + nt * 16 + lm;
            if (col >= N) continue;
            const float bv = bias ? bias[col] : 0.0f;
#pragma unroll
            for (int v = 0; v < 8; ++v) {
                const int row = blockIdx.y * 128 + mrow0 + am * 16 + hf * 8 + v;
                if (row >= M) continue;
                float val = alpha * acc[am][nt][v] + bv;
                const long idx = Cbase + (long)row * ldc + col;
                if (cF32) {
                    float* C = (float*)Cout;
                    if (resid) val += resid[idx];
                    C[idx] = val;
                } else {
                    ((bf16_t*)Cout)[idx] = f2bf(val);
                }
            }
        }
}

// ============================================================
// MoE FFN1: Hmid[slot,:] = gelu(x2[tok(slot)] @ w1t[e] + b1[e])
// ============================================================
__global__ __launch_bounds__(256)
void moe_ffn1(const bf16_t* __restrict__ X, const bf16_t* __restrict__ W1t,
              const float* __restrict__ b1, bf16_t* __restrict__ Hmid,
              const int* __restrict__ slotTok, const int* __restrict__ blockExpert,
              const int* __restrict__ nBlocks, int Hdim, int Dk)
{
    const int rb = blockIdx.y;
    if (rb >= *nBlocks) return;
    const int e = blockExpert[rb];
    const bf16_t* Bb = W1t + (long)e * Hdim * Dk + (long)blockIdx.x * 128 * Dk;

    v8f acc[2][4] = {};
    gemm_mainloop(X, Dk, Bb, Dk, Dk, 0, Hdim - blockIdx.x * 128,
                  slotTok + rb * 128, acc);

    const int tid = threadIdx.x, lane = tid & 31, wv = tid >> 5;
    const int hf = lane >> 4, lm = lane & 15;
    const int mrow0 = (wv & 3) * 32, ncol0 = (wv >> 2) * 64;
#pragma unroll
    for (int am = 0; am < 2; ++am)
#pragma unroll
        for (int nt = 0; nt < 4; ++nt) {
            const int col = blockIdx.x * 128 + ncol0 + nt * 16 + lm;
            const float bv = b1[(long)e * Hdim + col];
#pragma unroll
            for (int v = 0; v < 8; ++v) {
                const int slot = rb * 128 + mrow0 + am * 16 + hf * 8 + v;
                const float xv = acc[am][nt][v] + bv;
                const float g = 0.5f * xv * (1.0f + erff(xv * 0.70710678118654752f));
                Hmid[(long)slot * Hdim + col] = f2bf(g);
            }
        }
}

// ============================================================
// MoE FFN2: out[tok] += gate * (Hmid[slot] @ w2t[e] + b2[e])
// ============================================================
__global__ __launch_bounds__(256)
void moe_ffn2(const bf16_t* __restrict__ Hmid, const bf16_t* __restrict__ W2t,
              const float* __restrict__ b2, float* __restrict__ Out,
              const int* __restrict__ slotTok, const float* __restrict__ slotGate,
              const int* __restrict__ blockExpert, const int* __restrict__ nBlocks,
              int Dout, int Hk)
{
    const int rb = blockIdx.y;
    if (rb >= *nBlocks) return;
    const int e = blockExpert[rb];
    const bf16_t* Ab = Hmid + (long)rb * 128 * Hk;
    const bf16_t* Bb = W2t + (long)e * Dout * Hk + (long)blockIdx.x * 128 * Hk;

    v8f acc[2][4] = {};
    gemm_mainloop(Ab, Hk, Bb, Hk, Hk, 128, Dout - blockIdx.x * 128, nullptr, acc);

    const int tid = threadIdx.x, lane = tid & 31, wv = tid >> 5;
    const int hf = lane >> 4, lm = lane & 15;
    const int mrow0 = (wv & 3) * 32, ncol0 = (wv >> 2) * 64;
#pragma unroll
    for (int am = 0; am < 2; ++am)
#pragma unroll
        for (int nt = 0; nt < 4; ++nt) {
            const int col = blockIdx.x * 128 + ncol0 + nt * 16 + lm;
            if (col >= Dout) continue;
            const float bv = b2[(long)e * Dout + col];
#pragma unroll
            for (int v = 0; v < 8; ++v) {
                const int slot = rb * 128 + mrow0 + am * 16 + hf * 8 + v;
                const int tok = slotTok[slot];
                if (tok >= 0) {
                    const float val = slotGate[slot] * (acc[am][nt][v] + bv);
                    atomicAdd(&Out[(long)tok * Dout + col], val);
                }
            }
        }
}

// ============================================================
// LayerNorm (fp32 in -> bf16 out), one block per token
// ============================================================
__global__ __launch_bounds__(256)
void layernorm_bf16(const float* __restrict__ X, const float* __restrict__ g,
                    const float* __restrict__ b, bf16_t* __restrict__ Y, int Dd)
{
    const int t = blockIdx.x;
    const float* x = X + (long)t * Dd;
    __shared__ float sh[8];

    auto blockSum = [&](float v) -> float {
#pragma unroll
        for (int o = 16; o > 0; o >>= 1) v += __shfl_xor(v, o, 32);
        if ((threadIdx.x & 31) == 0) sh[threadIdx.x >> 5] = v;
        __syncthreads();
        float r = 0.f;
#pragma unroll
        for (int i = 0; i < 8; ++i) r += sh[i];
        __syncthreads();
        return r;
    };

    float s = 0.f;
    for (int i = threadIdx.x; i < Dd; i += 256) s += x[i];
    const float mean = blockSum(s) / Dd;
    float vs = 0.f;
    for (int i = threadIdx.x; i < Dd; i += 256) { float d = x[i] - mean; vs += d * d; }
    const float inv = rsqrtf(blockSum(vs) / Dd + 1e-5f);
    for (int i = threadIdx.x; i < Dd; i += 256)
        Y[(long)t * Dd + i] = f2bf((x[i] - mean) * inv * g[i] + b[i]);
}

// ============================================================
// Row softmax over scores (fp32) -> attn (bf16), with key mask
// ============================================================
__global__ __launch_bounds__(256)
void softmax_rows(const float* __restrict__ S, const unsigned char* __restrict__ mask,
                  bf16_t* __restrict__ A, int Scols)
{
    const long row = blockIdx.x;
    const float* src = S + row * Scols;
    const int bidx = (int)(row >> 14);        // row / (NH*S)
    const unsigned char* mrow = mask + (long)bidx * Scols;
    __shared__ float sh[8];

    auto blockRed = [&](float v, bool isMax) -> float {
#pragma unroll
        for (int o = 16; o > 0; o >>= 1) {
            float o2 = __shfl_xor(v, o, 32);
            v = isMax ? fmaxf(v, o2) : (v + o2);
        }
        if ((threadIdx.x & 31) == 0) sh[threadIdx.x >> 5] = v;
        __syncthreads();
        float r = isMax ? -3.4e38f : 0.f;
#pragma unroll
        for (int i = 0; i < 8; ++i) r = isMax ? fmaxf(r, sh[i]) : (r + sh[i]);
        __syncthreads();
        return r;
    };

    float m = -3.4e38f;
    for (int i = threadIdx.x; i < Scols; i += 256) {
        float v = mrow[i] ? src[i] : -1e30f;
        m = fmaxf(m, v);
    }
    m = blockRed(m, true);
    float s = 0.f;
    for (int i = threadIdx.x; i < Scols; i += 256) {
        float v = mrow[i] ? src[i] : -1e30f;
        s += expf(v - m);
    }
    s = blockRed(s, false);
    const float rinv = 1.0f / s;
    for (int i = threadIdx.x; i < Scols; i += 256) {
        float v = mrow[i] ? src[i] : -1e30f;
        A[row * Scols + i] = f2bf(expf(v - m) * rinv);
    }
}

// ============================================================
// Weight conversion / transpose
// ============================================================
__global__ void cvt_f32_bf16(const float* __restrict__ src, bf16_t* __restrict__ dst, long n)
{
    const long i = (long)blockIdx.x * 256 + threadIdx.x;
    if (i < n) dst[i] = f2bf(src[i]);
}

// dst[z][c][r] = src[z][r][c]   (block 32x8, tile 32x32)
__global__ void transpose_cvt(const float* __restrict__ src, bf16_t* __restrict__ dst,
                              int R, int C)
{
    __shared__ float tile[32][33];
    const long zoff = (long)blockIdx.z * R * C;
    const int r0 = blockIdx.y * 32, c0 = blockIdx.x * 32;
    const int tx = threadIdx.x, ty = threadIdx.y;
    for (int i = ty; i < 32; i += 8) {
        const int r = r0 + i, c = c0 + tx;
        tile[i][tx] = (r < R && c < C) ? src[zoff + (long)r * C + c] : 0.f;
    }
    __syncthreads();
    for (int i = ty; i < 32; i += 8) {
        const int c = c0 + i, r = r0 + tx;
        if (c < C && r < R) dst[zoff + (long)c * R + r] = f2bf(tile[tx][i]);
    }
}

// vt[(b*NH+h)][dh][s] = qkv[(b*S+s)][2D + h*DH + dh]
__global__ void make_vt(const bf16_t* __restrict__ qkv, bf16_t* __restrict__ vt)
{
    const long i = (long)blockIdx.x * 256 + threadIdx.x;
    if (i >= (long)CBH * CDH * CS) return;
    const int s  = (int)(i & (CS - 1));
    long r = i >> 10;
    const int dh = (int)(r & (CDH - 1)); r >>= 6;
    const int h  = (int)(r & (CNH - 1));
    const int b  = (int)(r >> 4);
    vt[i] = qkv[((long)(b * CS + s)) * (3 * CD) + 2 * CD + h * CDH + dh];
}

// ============================================================
// MoE routing
// ============================================================
__global__ void moe_init(int* counts, int* fill, float* sumDense, float* lse2,
                         int* slotTok, int nslots)
{
    const int i = blockIdx.x * 256 + threadIdx.x;
    if (i < nslots) slotTok[i] = -1;
    if (i < CE) { counts[i] = 0; fill[i] = 0; sumDense[i] = 0.f; }
    if (i == 0) lse2[0] = 0.f;
}

__global__ __launch_bounds__(256)
void router_kernel(const bf16_t* __restrict__ X2, const float* __restrict__ Wr,
                   int* __restrict__ topi, float* __restrict__ topw,
                   int* __restrict__ counts, float* __restrict__ sumDense,
                   float* __restrict__ lse2sum)
{
    const int wv = threadIdx.x >> 5, lane = threadIdx.x & 31;
    const int t = blockIdx.x * 8 + wv;
    if (t >= CT) return;
    float acc[CE];
#pragma unroll
    for (int e = 0; e < CE; ++e) acc[e] = 0.f;
    for (int d = lane; d < CD; d += 32) {
        const float xv = bf2f(X2[(long)t * CD + d]);
#pragma unroll
        for (int e = 0; e < CE; ++e) acc[e] += xv * Wr[e * CD + d];
    }
#pragma unroll
    for (int e = 0; e < CE; ++e)
#pragma unroll
        for (int o = 16; o > 0; o >>= 1) acc[e] += __shfl_xor(acc[e], o, 32);
    if (lane == 0) {
        int i0 = 0; float l0 = acc[0];
#pragma unroll
        for (int e = 1; e < CE; ++e) if (acc[e] > l0) { l0 = acc[e]; i0 = e; }
        int i1 = -1; float l1 = -3.4e38f;
#pragma unroll
        for (int e = 0; e < CE; ++e) if (e != i0 && acc[e] > l1) { l1 = acc[e]; i1 = e; }
        const float e1 = expf(l1 - l0);
        const float w0 = 1.0f / (1.0f + e1);
        topi[t * 2] = i0; topi[t * 2 + 1] = i1;
        topw[t * 2] = w0; topw[t * 2 + 1] = e1 * w0;
        atomicAdd(counts + i0, 1);
        atomicAdd(counts + i1, 1);
        float m = acc[0];
#pragma unroll
        for (int e = 1; e < CE; ++e) m = fmaxf(m, acc[e]);
        float s = 0.f; float pp[CE];
#pragma unroll
        for (int e = 0; e < CE; ++e) { pp[e] = expf(acc[e] - m); s += pp[e]; }
        const float rs = 1.0f / s;
#pragma unroll
        for (int e = 0; e < CE; ++e) atomicAdd(sumDense + e, pp[e] * rs);
        const float lse = m + logf(s);
        atomicAdd(lse2sum, lse * lse);
    }
}

__global__ void moe_scan(const int* counts, int* segBase, int* blockExpert,
                         int* nBlocks, const float* sumDense, const float* lse2sum,
                         float* outLoss)
{
    if (threadIdx.x != 0 || blockIdx.x != 0) return;
    int nb = 0;
    for (int e = 0; e < CE; ++e) {
        segBase[e] = nb * 128;
        const int be = (counts[e] + 127) >> 7;
        for (int k = 0; k < be; ++k) blockExpert[nb + k] = e;
        nb += be;
    }
    *nBlocks = nb;
    float laux = 0.f;
    for (int e = 0; e < CE; ++e) {
        const float frac = (float)counts[e] / (float)(CT * 2);
        laux += frac * (sumDense[e] / (float)CT);
    }
    outLoss[0] = 0.01f * (float)CE * laux;              // ALPHA * E * sum
    outLoss[1] = 0.001f * lse2sum[0] / (float)CT;       // CZ * mean(lse^2)
}

__global__ void moe_scatter(const int* __restrict__ topi, const float* __restrict__ topw,
                            const int* __restrict__ segBase, int* __restrict__ fill,
                            int* __restrict__ slotTok, float* __restrict__ slotGate)
{
    const int t = blockIdx.x * 256 + threadIdx.x;
    if (t >= CT) return;
#pragma unroll
    for (int k = 0; k < 2; ++k) {
        const int e = topi[t * 2 + k];
        const int p = atomicAdd(fill + e, 1);
        const int s = segBase[e] + p;
        slotTok[s] = t;
        slotGate[s] = topw[t * 2 + k];
    }
}

// ============================================================
// Host launch
// ============================================================
extern "C" void kernel_launch(void* const* d_in, const int* in_sizes, int n_in,
                              void* d_out, int out_size, void* d_ws, size_t ws_size,
                              hipStream_t stream)
{
    const float* x        = (const float*)d_in[0];
    const unsigned char* mask = (const unsigned char*)d_in[1];
    const float* ln1_g    = (const float*)d_in[2];
    const float* ln1_b    = (const float*)d_in[3];
    const float* ln2_g    = (const float*)d_in[4];
    const float* ln2_b    = (const float*)d_in[5];
    const float* w_qkv    = (const float*)d_in[6];
    const float* b_qkv    = (const float*)d_in[7];
    const float* w_o      = (const float*)d_in[8];
    const float* b_o      = (const float*)d_in[9];
    const float* w_router = (const float*)d_in[10];
    const float* w1       = (const float*)d_in[11];
    const float* b1       = (const float*)d_in[12];
    const float* w2       = (const float*)d_in[13];
    const float* b2       = (const float*)d_in[14];
    float* out = (float*)d_out;                 // [T*D out][l_aux][cz_lz]

    // ---- workspace carve-out ----
    char* p = (char*)d_ws;
    auto alloc = [&](size_t bytes) -> void* {
        void* r = (void*)p;
        p += (bytes + 255) & ~(size_t)255;
        return r;
    };
    bf16_t* hbf     = (bf16_t*)alloc((size_t)CT * CD * 2);            // ln1(x)
    bf16_t* wqkv_bf = (bf16_t*)alloc((size_t)3 * CD * CD * 2);
    bf16_t* wo_bf   = (bf16_t*)alloc((size_t)CD * CD * 2);
    bf16_t* w1t     = (bf16_t*)alloc((size_t)CE * CH * CD * 2);       // [E,H,D]
    bf16_t* w2t     = (bf16_t*)alloc((size_t)CE * CD * CH * 2);       // [E,D,H]
    bf16_t* qkv_bf  = (bf16_t*)alloc((size_t)CT * 3 * CD * 2);
    bf16_t* vt      = (bf16_t*)alloc((size_t)CBH * CDH * CS * 2);
    float*  scores  = (float*) alloc((size_t)CBH * CS * CS * 4);      // 128 MB (fits L2)
    bf16_t* attn    = (bf16_t*)alloc((size_t)CBH * CS * CS * 2);
    bf16_t* ao_bf   = (bf16_t*)alloc((size_t)CT * CD * 2);
    bf16_t* x2_bf   = (bf16_t*)alloc((size_t)CT * CD * 2);
    bf16_t* hmid    = (bf16_t*)alloc((size_t)NSLOTS * CH * 2);
    int*    topi    = (int*)   alloc((size_t)CT * 2 * 4);
    float*  topw    = (float*) alloc((size_t)CT * 2 * 4);
    int*    counts  = (int*)   alloc(CE * 4);
    int*    fill    = (int*)   alloc(CE * 4);
    int*    segBase = (int*)   alloc(CE * 4);
    int*    blockExpert = (int*)alloc(64 * 4);
    int*    nBlocks = (int*)   alloc(4);
    int*    slotTok = (int*)   alloc(NSLOTS * 4);
    float*  slotGate= (float*) alloc(NSLOTS * 4);
    float*  sumDense= (float*) alloc(CE * 4);
    float*  lse2    = (float*) alloc(4);

    // ---- 1. weights -> bf16 (w1/w2 transposed to NT layout) ----
    {
        long n = (long)3 * CD * CD;
        cvt_f32_bf16<<<dim3((unsigned)((n + 255) / 256)), 256, 0, stream>>>(w_qkv, wqkv_bf, n);
        n = (long)CD * CD;
        cvt_f32_bf16<<<dim3((unsigned)((n + 255) / 256)), 256, 0, stream>>>(w_o, wo_bf, n);
        transpose_cvt<<<dim3(CH / 32, CD / 32, CE), dim3(32, 8), 0, stream>>>(w1, w1t, CD, CH);
        transpose_cvt<<<dim3(CD / 32, CH / 32, CE), dim3(32, 8), 0, stream>>>(w2, w2t, CH, CD);
    }

    // ---- 2. LN1 ----
    layernorm_bf16<<<CT, 256, 0, stream>>>(x, ln1_g, ln1_b, hbf, CD);

    // ---- 3. QKV projection: [T,D] x [3D,D]^NT -> qkv bf16 [T,3D] ----
    gemm_nt_batched<<<dim3(3 * CD / 128, CT / 128, 1), 256, 0, stream>>>(
        hbf, CD, 0, 0, wqkv_bf, CD, 0, 0, b_qkv,
        qkv_bf, 0, 3 * CD, 0, 0, nullptr,
        CT, 3 * CD, CD, 1, 1.0f);

    // ---- 4. V^T for the AV GEMM ----
    make_vt<<<(CBH * CDH * CS) / 256, 256, 0, stream>>>(qkv_bf, vt);

    // ---- 5. scores = Q K^T / sqrt(DH)  (batched over B*NH heads, straight from qkv) ----
    gemm_nt_batched<<<dim3(CS / 128, CS / 128, CBH), 256, 0, stream>>>(
        qkv_bf,            3 * CD, (long)CS * 3 * CD, CDH,     // Q: b-stride, h-stride
        qkv_bf + CD,       3 * CD, (long)CS * 3 * CD, CDH,     // K
        nullptr,
        scores, 1, CS, (long)CNH * CS * CS, (long)CS * CS, nullptr,
        CS, CS, CDH, CNH, 0.125f);

    // ---- 6. masked softmax -> attn bf16 ----
    softmax_rows<<<CBH * CS, 256, 0, stream>>>(scores, mask, attn, CS);

    // ---- 7. AV: attn [S,S] x V^T [DH,S]^NT -> ao bf16 head-interleaved [B,S,D] ----
    gemm_nt_batched<<<dim3(1, CS / 128, CBH), 256, 0, stream>>>(
        attn, CS, (long)CNH * CS * CS, (long)CS * CS,
        vt,   CS, (long)CNH * CDH * CS, (long)CDH * CS,
        nullptr,
        ao_bf, 0, CD, (long)CS * CD, CDH, nullptr,
        CS, CDH, CS, CNH, 1.0f);

    // ---- 8. O projection + residual: x1 = x + ao@Wo^T + b_o  (writes d_out) ----
    gemm_nt_batched<<<dim3(CD / 128, CT / 128, 1), 256, 0, stream>>>(
        ao_bf, CD, 0, 0, wo_bf, CD, 0, 0, b_o,
        out, 1, CD, 0, 0, x,
        CT, CD, CD, 1, 1.0f);

    // ---- 9. LN2 on x1 ----
    layernorm_bf16<<<CT, 256, 0, stream>>>(out, ln2_g, ln2_b, x2_bf, CD);

    // ---- 10. routing ----
    moe_init<<<(NSLOTS + 255) / 256, 256, 0, stream>>>(counts, fill, sumDense, lse2,
                                                       slotTok, NSLOTS);
    router_kernel<<<CT / 8, 256, 0, stream>>>(x2_bf, w_router, topi, topw,
                                              counts, sumDense, lse2);
    moe_scan<<<1, 1, 0, stream>>>(counts, segBase, blockExpert, nBlocks,
                                  sumDense, lse2, out + (size_t)CT * CD);
    moe_scatter<<<(CT + 255) / 256, 256, 0, stream>>>(topi, topw, segBase, fill,
                                                      slotTok, slotGate);

    // ---- 11. sparse expert FFN (top-2 only; gates zero elsewhere) ----
    moe_ffn1<<<dim3(CH / 128, MAXRB), 256, 0, stream>>>(
        x2_bf, w1t, b1, hmid, slotTok, blockExpert, nBlocks, CH, CD);
    moe_ffn2<<<dim3(CD / 128, MAXRB), 256, 0, stream>>>(
        hmid, w2t, b2, out, slotTok, slotGate, blockExpert, nBlocks, CD, CH);
}
